// FAGCN_MOE_22110491640670
// MI455X (gfx1250) — matrix operand order repaired and checked by reading, not compiled, not run on hardware
//
#include <hip/hip_runtime.h>
#include <math.h>

#define Dm 96
#define Lm 64
#define HDm 12
#define NHm 8

typedef _Float16 v16h __attribute__((ext_vector_type(16)));
typedef float    v8f  __attribute__((ext_vector_type(8)));

__device__ __forceinline__ v8f wmma16(v16h a, v16h b, v8f c) {
    // D = A(16x32 f16) x B(32x16 f16) + C(16x16 f32)
    return __builtin_amdgcn_wmma_f32_16x16x32_f16(false, a, false, b, (short)0, c, false, false);
}

// ---------------- generic helpers ----------------
__global__ void zero_kernel(float* __restrict__ p, int n) {
    int i = blockIdx.x * blockDim.x + threadIdx.x;
    if (i < n) p[i] = 0.f;
}

__global__ void f32_to_f16_kernel(const float* __restrict__ src, _Float16* __restrict__ dst,
                                  int n) {
    int i = blockIdx.x * blockDim.x + threadIdx.x;
    if (i < n) dst[i] = (_Float16)src[i];
}

// ---------------- GNN kernels ----------------
__global__ void deg_kernel(const int* __restrict__ erow, float* __restrict__ deg, int E) {
    int e = blockIdx.x * blockDim.x + threadIdx.x;
    if (e < E) atomicAdd(&deg[erow[e]], 1.0f);
}

__global__ void dinv_kernel(float* __restrict__ deg, int n) {
    int i = blockIdx.x * blockDim.x + threadIdx.x;
    if (i < n) { float d = deg[i]; deg[i] = (d > 0.f) ? rsqrtf(d) : 0.f; }
}

// per-row dots gl = x.al, gr = x.ar ; one wave32 per row
__global__ void glgr_kernel(const float* __restrict__ x, const float* __restrict__ al,
                            const float* __restrict__ ar, float* __restrict__ gl,
                            float* __restrict__ gr, int N) {
    int lane = threadIdx.x & 31;
    int row  = blockIdx.x * 4 + (threadIdx.x >> 5);
    if (row >= N) return;
    float sl = 0.f, sr = 0.f;
    for (int k = lane; k < Dm; k += 32) {
        float xv = x[row * Dm + k];
        sl += xv * al[k];
        sr += xv * ar[k];
    }
    for (int off = 16; off; off >>= 1) {
        sl += __shfl_xor(sl, off, 32);
        sr += __shfl_xor(sr, off, 32);
    }
    if (lane == 0) { gl[row] = sl; gr[row] = sr; }
}

// edge scatter: agg[row,:] += coef * x[col,:]  (96 lanes per edge, 4 edges per block)
__global__ __launch_bounds__(384) void scatter_kernel(
        const int* __restrict__ erow, const int* __restrict__ ecol,
        const float* __restrict__ x, const float* __restrict__ gl,
        const float* __restrict__ gr, const float* __restrict__ dinv,
        float* __restrict__ agg, int E) {
    __shared__ float scoef[4];
    int e = blockIdx.x * 4 + threadIdx.y;
    bool valid = e < E;
    int r = 0, c = 0;
    if (valid) {
        r = erow[e]; c = ecol[e];
        if (threadIdx.x == 0)
            scoef[threadIdx.y] = tanhf(gl[r] + gr[c]) * dinv[r] * dinv[c];
    }
    __syncthreads();
    if (valid) {
        float coef = scoef[threadIdx.y];
        int d = threadIdx.x;
        atomicAdd(&agg[r * Dm + d], coef * x[c * Dm + d]);
    }
}

// x_out = relu(EPS * x0 + agg)
__global__ void combine_kernel(const float* __restrict__ x0, const float* __restrict__ agg,
                               float* __restrict__ xout, int n) {
    int i = blockIdx.x * blockDim.x + threadIdx.x;
    if (i < n) xout[i] = fmaxf(0.3f * x0[i] + agg[i], 0.f);
}

// ---------------- gating + concat (f16 output for WMMA GEMM) ----------------
__global__ void gate_kernel(const float* __restrict__ x, const float* __restrict__ w_ga,
                            const float* __restrict__ b_ga, const float* __restrict__ w_gb,
                            const float* __restrict__ b_gb, const float* __restrict__ s_out,
                            const float* __restrict__ a_out, const float* __restrict__ b_out,
                            _Float16* __restrict__ catb, int N) {
    int lane = threadIdx.x & 31;
    int row  = blockIdx.x * 4 + (threadIdx.x >> 5);
    if (row >= N) return;
    float l0 = 0.f, l1 = 0.f, m0 = 0.f, m1 = 0.f;
    for (int k = lane; k < Dm; k += 32) {
        float xv = x[row * Dm + k];
        l0 += xv * w_ga[k];       l1 += xv * w_ga[Dm + k];
        m0 += xv * w_gb[k];       m1 += xv * w_gb[Dm + k];
    }
    for (int off = 16; off; off >>= 1) {
        l0 += __shfl_xor(l0, off, 32); l1 += __shfl_xor(l1, off, 32);
        m0 += __shfl_xor(m0, off, 32); m1 += __shfl_xor(m1, off, 32);
    }
    l0 += b_ga[0]; l1 += b_ga[1]; m0 += b_gb[0]; m1 += b_gb[1];
    float mx = fmaxf(l0, l1);
    float e0 = expf(l0 - mx), e1 = expf(l1 - mx);
    float ga0 = e0 / (e0 + e1), ga1 = e1 / (e0 + e1);
    mx = fmaxf(m0, m1);
    e0 = expf(m0 - mx); e1 = expf(m1 - mx);
    float gb0 = e0 / (e0 + e1), gb1 = e1 / (e0 + e1);
    for (int k = lane; k < Dm; k += 32) {
        float sv = s_out[row * Dm + k];
        catb[row * (2 * Dm) + k]      = (_Float16)(ga0 * a_out[row * Dm + k] + ga1 * sv);
        catb[row * (2 * Dm) + Dm + k] = (_Float16)(gb0 * b_out[row * Dm + k] + gb1 * sv);
    }
}

// ---------------- h = relu(cat @ w_lin.T + b_lin) via WMMA ----------------
// M = N (mult of 16), N-dim = 96 (6 tiles), K = 192 (6 steps of 32). 6 waves/block.
// Writes both f32 h (for q0) and f16 h (for K/V gather).
__global__ __launch_bounds__(192) void lin_kernel(const _Float16* __restrict__ catb,
                                                  const _Float16* __restrict__ w_lin_h,
                                                  const float* __restrict__ b_lin,
                                                  float* __restrict__ hbuf,
                                                  _Float16* __restrict__ hbuf_h, int N) {
    int lane = threadIdx.x & 31;
    int nt   = threadIdx.x >> 5;          // 0..5
    int mbase = blockIdx.x * 16;
    int row = mbase + (lane & 15);
    if (row >= N) row = N - 1;            // select, keeps EXEC full
    int n = nt * 16 + (lane & 15);
    v8f acc = {0.f, 0.f, 0.f, 0.f, 0.f, 0.f, 0.f, 0.f};
    for (int ks = 0; ks < 6; ++ks) {
        v16h a, b;
        int ka = ks * 32 + ((lane & 16) ? 16 : 0);
#pragma unroll
        for (int j = 0; j < 16; ++j) a[j] = catb[row * 192 + ka + j];
        int kb = ks * 32 + ((lane & 16) ? 16 : 0);
#pragma unroll
        for (int j = 0; j < 16; ++j) b[j] = w_lin_h[n * 192 + kb + j];
        acc = wmma16(a, b, acc);
    }
    float bias = b_lin[n];
    int mrow = mbase + ((lane & 16) ? 8 : 0);
    if (mbase + 16 <= N) {                // uniform: full tile fast path
#pragma unroll
        for (int r = 0; r < 8; ++r) {
            float v = fmaxf(acc[r] + bias, 0.f);
            hbuf[(mrow + r) * Dm + n]   = v;
            hbuf_h[(mrow + r) * Dm + n] = (_Float16)v;
        }
    } else {
#pragma unroll
        for (int r = 0; r < 8; ++r) {
            int m = mrow + r;
            if (m < N) {
                float v = fmaxf(acc[r] + bias, 0.f);
                hbuf[m * Dm + n]   = v;
                hbuf_h[m * Dm + n] = (_Float16)v;
            }
        }
    }
}

// ---------------- fused per-group attention ----------------
// Only q at position 0 is needed. Per group: K,V = tgt @ Wk.T / Wv.T via WMMA
// (M=64, N=96, K=96), then scores/softmax/ctx/out-proj in f32.
__global__ __launch_bounds__(256) void attn_kernel(const float* __restrict__ hbuf,
                                                   const _Float16* __restrict__ hbuf_h,
                                                   const int* __restrict__ tids,
                                                   const int* __restrict__ tlens,
                                                   const float* __restrict__ w_qkv,
                                                   const _Float16* __restrict__ w_qkv_h,
                                                   const float* __restrict__ b_qkv,
                                                   const float* __restrict__ w_o,
                                                   const float* __restrict__ b_o,
                                                   float* __restrict__ out_vec,
                                                   float* __restrict__ out_w) {
    __shared__ _Float16 sT[Lm * Dm];   // tgt rows f16
    __shared__ float    sK[Lm * Dm];
    __shared__ float    sV[Lm * Dm];
    __shared__ float    sQ[Dm];
    __shared__ float    sA[NHm * Lm];
    __shared__ float    sC[Dm];        // phase0/1: tgt row0 f32 ; phase4: ctx

    const int g    = blockIdx.x;
    const int tid  = threadIdx.x;
    const int lane = tid & 31;
    const int wave = tid >> 5;

    // phase 0: load target rows (f16 direct) and row 0 (f32)
    for (int idx = tid; idx < Lm * Dm; idx += 256) {
        int l = idx / Dm, k = idx - l * Dm;
        int rowid = tids[g * Lm + l];
        sT[idx] = hbuf_h[rowid * Dm + k];
    }
    if (tid < Dm) {
        int rowid0 = tids[g * Lm];
        sC[tid] = hbuf[rowid0 * Dm + tid];
    }
    __syncthreads();

    // phase 1a: q0 (f32) = tgt0 @ Wq.T + b, scaled by 1/sqrt(DH)
    if (tid < Dm) {
        float acc = b_qkv[tid];
        for (int k = 0; k < Dm; ++k) acc += sC[k] * w_qkv[tid * Dm + k];
        sQ[tid] = acc * 0.28867513459481287f;  // 1/sqrt(12)
    }

    // phase 1b: K and V via WMMA — 48 tiles (24 K + 24 V), 6 per wave
    for (int tile = wave; tile < 48; tile += 8) {
        int isV = tile >= 24;
        int t24 = isV ? (tile - 24) : tile;
        int mt = t24 / 6, nt = t24 - mt * 6;
        int row = mt * 16 + (lane & 15);
        int n   = nt * 16 + (lane & 15);
        int wrow = ((isV ? 2 : 1) * Dm + n) * Dm;   // w_qkv row for this output col
        v8f acc = {0.f, 0.f, 0.f, 0.f, 0.f, 0.f, 0.f, 0.f};
        for (int ks = 0; ks < 3; ++ks) {
            v16h a, b;
            int ka = ks * 32 + ((lane & 16) ? 16 : 0);
#pragma unroll
            for (int j = 0; j < 16; ++j) a[j] = sT[row * Dm + ka + j];
            int kb = ks * 32 + ((lane & 16) ? 16 : 0);
#pragma unroll
            for (int j = 0; j < 16; ++j) b[j] = w_qkv_h[wrow + kb + j];
            acc = wmma16(a, b, acc);
        }
        float bias = b_qkv[(isV ? 2 : 1) * Dm + n];
        float* dst = isV ? sV : sK;
#pragma unroll
        for (int r = 0; r < 8; ++r) {
            int m = mt * 16 + r + ((lane & 16) ? 8 : 0);
            dst[m * Dm + n] = acc[r] + bias;
        }
    }
    __syncthreads();

    // phase 2: masked scores (h, m)
    int len = tlens[g];
    for (int idx = tid; idx < NHm * Lm; idx += 256) {
        int hh = idx / Lm, m = idx - hh * Lm;
        float s;
        if (m < len) {
            s = 0.f;
            for (int d2 = 0; d2 < HDm; ++d2)
                s += sQ[hh * HDm + d2] * sK[m * Dm + hh * HDm + d2];
        } else {
            s = -INFINITY;
        }
        sA[idx] = s;
    }
    __syncthreads();

    // phase 3: softmax per head
    if (tid < NHm) {
        float mx = -INFINITY;
        for (int m = 0; m < Lm; ++m) mx = fmaxf(mx, sA[tid * Lm + m]);
        float sum = 0.f;
        for (int m = 0; m < Lm; ++m) {
            float e = expf(sA[tid * Lm + m] - mx);
            sA[tid * Lm + m] = e;
            sum += e;
        }
        float inv = 1.f / sum;
        for (int m = 0; m < Lm; ++m) sA[tid * Lm + m] *= inv;
    }
    __syncthreads();

    // phase 4: ctx (tid<96) and head-mean weights (tid 128..191)
    if (tid < Dm) {
        int hh = tid / HDm;
        float c = 0.f;
        for (int m = 0; m < Lm; ++m) c += sA[hh * Lm + m] * sV[m * Dm + tid];
        sC[tid] = c;
    } else if (tid >= 128 && tid < 128 + Lm) {
        int m = tid - 128;
        float s = 0.f;
        for (int hh = 0; hh < NHm; ++hh) s += sA[hh * Lm + m];
        out_w[g * Lm + m] = s * (1.0f / NHm);
    }
    __syncthreads();

    // phase 5: vectors = ctx @ w_o.T + b_o
    if (tid < Dm) {
        float acc = b_o[tid];
        for (int j = 0; j < Dm; ++j) acc += sC[j] * w_o[tid * Dm + j];
        out_vec[g * Dm + tid] = acc;
    }
}

// ---------------- launch ----------------
extern "C" void kernel_launch(void* const* d_in, const int* in_sizes, int n_in,
                              void* d_out, int out_size, void* d_ws, size_t ws_size,
                              hipStream_t stream) {
    const float* x     = (const float*)d_in[0];
    const float* a_l   = (const float*)d_in[1];
    const float* a_r   = (const float*)d_in[2];
    const float* w_ga  = (const float*)d_in[3];
    const float* b_ga  = (const float*)d_in[4];
    const float* w_gb  = (const float*)d_in[5];
    const float* b_gb  = (const float*)d_in[6];
    const float* w_lin = (const float*)d_in[7];
    const float* b_lin = (const float*)d_in[8];
    const float* w_qkv = (const float*)d_in[9];
    const float* b_qkv = (const float*)d_in[10];
    const float* w_o   = (const float*)d_in[11];
    const float* b_o   = (const float*)d_in[12];
    const int*   g0    = (const int*)d_in[13];
    const int*   g1    = (const int*)d_in[14];
    const int*   g2    = (const int*)d_in[15];
    const int*   tids  = (const int*)d_in[16];
    const int*   tlens = (const int*)d_in[17];

    const int N = in_sizes[0] / Dm;       // 50000
    const int E = in_sizes[13] / 2;       // 800000
    const int G = in_sizes[17];           // 2048
    const int ND = N * Dm;
    float* out = (float*)d_out;

    // workspace carving (256B aligned)
    char* wp = (char*)d_ws;
    auto carve = [&](size_t bytes) -> char* {
        char* p = wp;
        wp += (bytes + 255) & ~(size_t)255;
        return p;
    };
    float* deg   = (float*)carve((size_t)N * 4);
    float* gl    = (float*)carve((size_t)N * 4);
    float* gr    = (float*)carve((size_t)N * 4);
    float* agg   = (float*)carve((size_t)ND * 4);
    float* t1    = (float*)carve((size_t)ND * 4);
    float* gout0 = (float*)carve((size_t)ND * 4);   // s_out (g0)
    float* gout1 = (float*)carve((size_t)ND * 4);   // a_out (g1)
    float* gout2 = (float*)carve((size_t)ND * 4);   // b_out (g2)
    _Float16* catb     = (_Float16*)carve((size_t)N * 2 * Dm * 2);
    float*    hbuf     = (float*)carve((size_t)ND * 4);
    _Float16* hbuf_h   = (_Float16*)carve((size_t)ND * 2);
    _Float16* w_lin_h  = (_Float16*)carve((size_t)Dm * 2 * Dm * 2);     // 96x192
    _Float16* w_qkv_h  = (_Float16*)carve((size_t)3 * Dm * Dm * 2);     // 288x96
    (void)ws_size; (void)n_in; (void)out_size;

    // one-time (per launch) weight conversions to f16
    f32_to_f16_kernel<<<(Dm * 2 * Dm + 255) / 256, 256, 0, stream>>>(w_lin, w_lin_h,
                                                                     Dm * 2 * Dm);
    f32_to_f16_kernel<<<(3 * Dm * Dm + 255) / 256, 256, 0, stream>>>(w_qkv, w_qkv_h,
                                                                     3 * Dm * Dm);

    const int* graphs[3] = {g0, g1, g2};
    float* gouts[3] = {gout0, gout1, gout2};

    for (int gi = 0; gi < 3; ++gi) {
        const int* erow = graphs[gi];
        const int* ecol = graphs[gi] + E;
        zero_kernel<<<(N + 255) / 256, 256, 0, stream>>>(deg, N);
        deg_kernel<<<(E + 255) / 256, 256, 0, stream>>>(erow, deg, E);
        dinv_kernel<<<(N + 255) / 256, 256, 0, stream>>>(deg, N);

        const float* xin = x;
        float* xout = t1;
        for (int layer = 0; layer < 2; ++layer) {
            glgr_kernel<<<(N + 3) / 4, 128, 0, stream>>>(xin, a_l + layer * Dm,
                                                         a_r + layer * Dm, gl, gr, N);
            zero_kernel<<<(ND + 255) / 256, 256, 0, stream>>>(agg, ND);
            scatter_kernel<<<(E + 3) / 4, dim3(96, 4), 0, stream>>>(erow, ecol, xin, gl, gr,
                                                                    deg, agg, E);
            combine_kernel<<<(ND + 255) / 256, 256, 0, stream>>>(x, agg, xout, ND);
            xin = t1;
            xout = gouts[gi];
        }
    }

    gate_kernel<<<(N + 3) / 4, 128, 0, stream>>>(x, w_ga, b_ga, w_gb, b_gb,
                                                 gout0, gout1, gout2, catb, N);
    lin_kernel<<<(N + 15) / 16, 192, 0, stream>>>(catb, w_lin_h, b_lin, hbuf, hbuf_h, N);
    attn_kernel<<<G, 256, 0, stream>>>(hbuf, hbuf_h, tids, tlens, w_qkv, w_qkv_h, b_qkv,
                                       w_o, b_o, out, out + (size_t)G * Dm);
}
